// MultiheadAttention_57028575756727
// MI455X (gfx1250) — matrix operand order repaired and checked by reading, not compiled, not run on hardware
//
#include <hip/hip_runtime.h>
#include <hip/hip_bf16.h>

// ---------------------------------------------------------------------------
// MI455X (gfx1250) attention block, bf16 WMMA everywhere.
//   B=4, S=2048, D=1024, dqk=dv=1024 (reference never splits heads).
// Pipeline:
//   1) gemm_xw_bf16 x3 : Q/K (row-major bf16), V (transposed [b][col][s] bf16)
//   2) cvt_f32_bf16    : WO -> bf16
//   3) attn_kernel     : 32-query strip per WG; scores (32x2048 f32) live in
//                        LDS (256KB) + Q strip (64KB) = 320KB LDS per WGP.
//                        Q strip staged with GLOBAL_LOAD_ASYNC_TO_LDS_B128.
//   4) outproj_ln      : O @ WO^T + residual + LayerNorm -> f32 out
// ---------------------------------------------------------------------------

typedef __attribute__((ext_vector_type(16))) __bf16 v16bf;
typedef __attribute__((ext_vector_type(8)))  float  v8f;

union FragU {
  v16bf v;
  uint4 q[2];
  unsigned u[8];
};

__device__ __forceinline__ v8f zero8() {
  v8f z;
#pragma unroll
  for (int i = 0; i < 8; ++i) z[i] = 0.0f;
  return z;
}

// round-to-nearest-even f32 -> bf16 pack (a -> low 16, b -> high 16)
__device__ __forceinline__ unsigned pk_bf16(float a, float b) {
  unsigned ua = __builtin_bit_cast(unsigned, a);
  unsigned ub = __builtin_bit_cast(unsigned, b);
  ua += 0x7FFFu + ((ua >> 16) & 1u);
  ub += 0x7FFFu + ((ub >> 16) & 1u);
  return (ua >> 16) | (ub & 0xFFFF0000u);
}
__device__ __forceinline__ unsigned short bf16_1(float a) {
  unsigned ua = __builtin_bit_cast(unsigned, a);
  ua += 0x7FFFu + ((ua >> 16) & 1u);
  return (unsigned short)(ua >> 16);
}

__device__ __forceinline__ v8f wmma_bf16(const FragU& a, const FragU& b, v8f c) {
  return __builtin_amdgcn_wmma_f32_16x16x32_bf16(
      false, a.v, false, b.v, (short)0, c, false, false);
}

// CDNA5 async global->LDS copy (16B per lane), tracked by ASYNCcnt.
// lds_off is the workgroup-LDS byte offset (our dynamic LDS starts at 0).
__device__ __forceinline__ void async_ld_lds_b128(unsigned lds_off,
                                                  const void* gaddr) {
  asm volatile("global_load_async_to_lds_b128 %0, %1, off"
               :: "v"(lds_off), "v"(gaddr)
               : "memory");
}
__device__ __forceinline__ void wait_asynccnt0() {
  asm volatile("s_wait_asynccnt 0x0" ::: "memory");
}
// Optimization fence: stops LICM/CSE from caching LDS fragments across
// outer-loop iterations (round-2 asm showed 512-VGPR hoist + scratch spills).
__device__ __forceinline__ void opt_fence() { asm volatile("" ::: "memory"); }

static constexpr int Sq = 2048;
static constexpr int NB = 4;
static constexpr int Dm = 1024;
static constexpr int Mrows = NB * Sq;  // 8192

// ---------------------------------------------------------------------------
// Kernel 1: C[M,1024](bf16) = X[M,1024](f32) @ W[1024,1024]^T (f32)
// Block tile 128x128, 8 waves, wave tile 64x32, K-step 32 via LDS.
// TR=true stores per-batch transposed: Out[(b*1024+col)*2048 + s]
// ---------------------------------------------------------------------------
template <bool TR>
__global__ __launch_bounds__(256)
void gemm_xw_bf16(const float* __restrict__ X, const float* __restrict__ W,
                  unsigned short* __restrict__ Out) {
  constexpr int K = 1024;
  constexpr int LDT = 40;  // u16 row stride (80B = 20 banks -> conflict free)
  __shared__ unsigned short As[128 * LDT];
  __shared__ unsigned short Ws[128 * LDT];

  const int t = threadIdx.x;
  const int wave = t >> 5, lane = t & 31;
  const int ln = lane & 15, h = lane >> 4;
  const int m0 = blockIdx.y * 128;
  const int n0 = blockIdx.x * 128;
  const int wm = wave >> 2;  // 0..1  (64 rows each)
  const int wn = wave & 3;   // 0..3  (32 cols each)

  v8f acc[4][2];
#pragma unroll
  for (int i = 0; i < 4; ++i)
#pragma unroll
    for (int j = 0; j < 2; ++j) acc[i][j] = zero8();

  for (int k0 = 0; k0 < K; k0 += 32) {
    __syncthreads();
#pragma unroll
    for (int i = 0; i < 4; ++i) {
      int slot = t + 256 * i;  // 1024 slots x 4 floats
      int row = slot >> 3;
      int c4 = (slot & 7) * 4;
      float4 av = *(const float4*)(X + (size_t)(m0 + row) * K + k0 + c4);
      uint2 pa; pa.x = pk_bf16(av.x, av.y); pa.y = pk_bf16(av.z, av.w);
      *(uint2*)(&As[row * LDT + c4]) = pa;
      float4 wv = *(const float4*)(W + (size_t)(n0 + row) * K + k0 + c4);
      uint2 pw; pw.x = pk_bf16(wv.x, wv.y); pw.y = pk_bf16(wv.z, wv.w);
      *(uint2*)(&Ws[row * LDT + c4]) = pw;
    }
    __syncthreads();

    FragU a[4], bf[2];
#pragma unroll
    for (int mf = 0; mf < 4; ++mf) {
      const unsigned short* p = &As[(wm * 64 + mf * 16 + ln) * LDT];
      a[mf].q[0] = *(const uint4*)(p + h * 8);       // K = h*8 .. h*8+7
      a[mf].q[1] = *(const uint4*)(p + 16 + h * 8);  // K = 16+h*8 ..
    }
#pragma unroll
    for (int nf = 0; nf < 2; ++nf) {
      const unsigned short* p = &Ws[(wn * 32 + nf * 16 + ln) * LDT];
      bf[nf].q[0] = *(const uint4*)(p + h * 8);       // K = h*8 .. h*8+7
      bf[nf].q[1] = *(const uint4*)(p + 16 + h * 8);  // K = 16+h*8 ..
    }
#pragma unroll
    for (int mf = 0; mf < 4; ++mf)
#pragma unroll
      for (int nf = 0; nf < 2; ++nf)
        acc[mf][nf] = wmma_bf16(a[mf], bf[nf], acc[mf][nf]);
  }

#pragma unroll
  for (int mf = 0; mf < 4; ++mf)
#pragma unroll
    for (int nf = 0; nf < 2; ++nf) {
      int col = n0 + wn * 32 + nf * 16 + ln;
#pragma unroll
      for (int v = 0; v < 8; ++v) {
        int row = m0 + wm * 64 + mf * 16 + v + 8 * h;
        unsigned short bb = bf16_1(acc[mf][nf][v]);
        if constexpr (!TR) {
          Out[(size_t)row * 1024 + col] = bb;
        } else {
          int b = row >> 11, s = row & 2047;
          Out[((size_t)(b * 1024 + col) << 11) + s] = bb;
        }
      }
    }
}

// ---------------------------------------------------------------------------
// Kernel 2: elementwise f32 -> bf16 (for WO)
// ---------------------------------------------------------------------------
__global__ __launch_bounds__(256)
void cvt_f32_bf16(const float* __restrict__ in, unsigned short* __restrict__ out,
                  int n) {
  int i = blockIdx.x * 256 + threadIdx.x;
  if (i < n) out[i] = bf16_1(in[i]);
}

// ---------------------------------------------------------------------------
// Kernel 3: attention for a 32-query strip of one batch.
//   LDS: sQ [32][1024] bf16 (64KB, async-staged) + sS [32][2048] f32 (256KB).
//   sInv[32] (per-row 1/sum) overlays the dead Q strip after phase 1.
// ---------------------------------------------------------------------------
__global__ __launch_bounds__(256)
void attn_kernel(const unsigned short* __restrict__ Qb,
                 const unsigned short* __restrict__ Kb,
                 const unsigned short* __restrict__ Vt,
                 const int* __restrict__ mask,
                 unsigned short* __restrict__ Ob) {
  extern __shared__ char smem[];
  unsigned short* sQ = (unsigned short*)smem;  // 64 KB @ LDS offset 0
  float* sS = (float*)(smem + 32 * 1024 * 2);  // 256 KB @ LDS offset 65536
  float* sInv = (float*)smem;                  // 128 B, reuses dead sQ

  const int t = threadIdx.x;
  const int wave = t >> 5, lane = t & 31;
  const int ln = lane & 15, h = lane >> 4;
  const int qb = blockIdx.x * 32;
  const int b = blockIdx.y;
  const size_t rowBase = (size_t)b * Sq + qb;

  {  // Phase 0: Q strip -> LDS via async DMA (64KB = 4096 x 16B)
    const char* src = (const char*)(Qb + rowBase * 1024);
#pragma unroll
    for (int i = 0; i < 16; ++i) {
      unsigned off = (unsigned)(t + 256 * i) * 16u;
      async_ld_lds_b128(off, src + off);
    }
    wait_asynccnt0();
  }
  __syncthreads();

  // Phase 1: scores = (Q K^T) * 1/sqrt(64) into sS; wave owns 256 keys
  const float scale = 0.125f;
  const int keyBase = wave * 256;
#pragma clang loop unroll(disable)
  for (int kt = 0; kt < 8; ++kt) {
    opt_fence();  // force per-kt re-read of Q frags from LDS (no spill-cache)
    v8f acc[2][2];
#pragma unroll
    for (int i = 0; i < 2; ++i)
#pragma unroll
      for (int j = 0; j < 2; ++j) acc[i][j] = zero8();
    for (int kf = 0; kf < 32; ++kf) {
      FragU a[2], bb[2];
#pragma unroll
      for (int mf = 0; mf < 2; ++mf) {
        const unsigned short* p = sQ + (mf * 16 + ln) * 1024 + kf * 32;
        a[mf].q[0] = *(const uint4*)(p + h * 8);
        a[mf].q[1] = *(const uint4*)(p + 16 + h * 8);
      }
#pragma unroll
      for (int nf = 0; nf < 2; ++nf) {
        int key = keyBase + kt * 32 + nf * 16 + ln;
        const unsigned short* p =
            Kb + ((size_t)b * Sq + key) * 1024 + kf * 32;
        bb[nf].q[0] = *(const uint4*)(p + h * 8);
        bb[nf].q[1] = *(const uint4*)(p + 16 + h * 8);
      }
#pragma unroll
      for (int mf = 0; mf < 2; ++mf)
#pragma unroll
        for (int nf = 0; nf < 2; ++nf)
          acc[mf][nf] = wmma_bf16(a[mf], bb[nf], acc[mf][nf]);
    }
#pragma unroll
    for (int mf = 0; mf < 2; ++mf)
#pragma unroll
      for (int nf = 0; nf < 2; ++nf) {
        int col = keyBase + kt * 32 + nf * 16 + ln;
#pragma unroll
        for (int v = 0; v < 8; ++v)
          sS[(mf * 16 + v + 8 * h) * 2048 + col] = acc[mf][nf][v] * scale;
      }
  }
  __syncthreads();

  // Phase 2: mask + softmax (unnormalized exp left in sS; 1/sum -> sInv)
  for (int rr = 0; rr < 4; ++rr) {
    int r = wave * 4 + rr;
    const int* mrow = mask + ((size_t)b * Sq + qb + r) * Sq;
    if (rr < 3) {  // stream next mask row ahead of the VALU loop
      const char* mnext = (const char*)(mrow + Sq);
      __builtin_prefetch(mnext + lane * 256, 0, 0);
      __builtin_prefetch(mnext + lane * 256 + 128, 0, 0);
    }
    float* srow = sS + r * 2048;
    float mx = -3.0e38f;
    for (int i = 0; i < 64; ++i) {
      int k = lane + 32 * i;
      float s = srow[k];
      s = (mrow[k] == 1) ? -1.0e9f : s;
      srow[k] = s;
      mx = fmaxf(mx, s);
    }
#pragma unroll
    for (int off = 16; off > 0; off >>= 1) mx = fmaxf(mx, __shfl_xor(mx, off, 32));
    float sum = 0.0f;
    for (int i = 0; i < 64; ++i) {
      int k = lane + 32 * i;
      float e = __expf(srow[k] - mx);
      srow[k] = e;
      sum += e;
    }
#pragma unroll
    for (int off = 16; off > 0; off >>= 1) sum += __shfl_xor(sum, off, 32);
    if (lane == 0) sInv[r] = 1.0f / sum;
  }
  __syncthreads();

  // Phase 3: O = P @ V; wave owns cols [wave*128, +128); 1/sum folded at end
  v8f acc[2][8];
#pragma unroll
  for (int i = 0; i < 2; ++i)
#pragma unroll
    for (int j = 0; j < 8; ++j) acc[i][j] = zero8();

#pragma clang loop unroll(disable)
  for (int kf = 0; kf < 64; ++kf) {  // key frags of 32
    opt_fence();
    FragU a[2];
#pragma unroll
    for (int mf = 0; mf < 2; ++mf) {
      const float* p = sS + (mf * 16 + ln) * 2048 + kf * 32;
      float4 f0 = *(const float4*)(p + h * 8);
      float4 f1 = *(const float4*)(p + h * 8 + 4);
      float4 f2 = *(const float4*)(p + 16 + h * 8);
      float4 f3 = *(const float4*)(p + 16 + h * 8 + 4);
      a[mf].u[0] = pk_bf16(f0.x, f0.y); a[mf].u[1] = pk_bf16(f0.z, f0.w);
      a[mf].u[2] = pk_bf16(f1.x, f1.y); a[mf].u[3] = pk_bf16(f1.z, f1.w);
      a[mf].u[4] = pk_bf16(f2.x, f2.y); a[mf].u[5] = pk_bf16(f2.z, f2.w);
      a[mf].u[6] = pk_bf16(f3.x, f3.y); a[mf].u[7] = pk_bf16(f3.z, f3.w);
    }
#pragma unroll
    for (int nf = 0; nf < 8; ++nf) {
      FragU bb;
      int col = wave * 128 + nf * 16 + ln;
      const unsigned short* p =
          Vt + (size_t)(b * 1024 + col) * Sq + kf * 32;
      bb.q[0] = *(const uint4*)(p + h * 8);
      bb.q[1] = *(const uint4*)(p + 16 + h * 8);
#pragma unroll
      for (int mf = 0; mf < 2; ++mf) acc[mf][nf] = wmma_bf16(a[mf], bb, acc[mf][nf]);
    }
  }

  float rinv[2][8];
#pragma unroll
  for (int mf = 0; mf < 2; ++mf)
#pragma unroll
    for (int v = 0; v < 8; ++v) rinv[mf][v] = sInv[mf * 16 + v + 8 * h];

#pragma unroll
  for (int mf = 0; mf < 2; ++mf)
#pragma unroll
    for (int nf = 0; nf < 8; ++nf) {
      int col = wave * 128 + nf * 16 + ln;
#pragma unroll
      for (int v = 0; v < 8; ++v) {
        int row = qb + mf * 16 + v + 8 * h;
        Ob[((size_t)b * Sq + row) * 1024 + col] =
            bf16_1(acc[mf][nf][v] * rinv[mf][v]);
      }
    }
}

// ---------------------------------------------------------------------------
// Kernel 4: proj = O @ WO^T ; out = LayerNorm(proj + inputQ)
//   LDS: sA [32][1024] bf16 (64KB, async-staged) + sP [32][1024] f32 (128KB).
// ---------------------------------------------------------------------------
__global__ __launch_bounds__(256)
void outproj_ln(const unsigned short* __restrict__ Ob,
                const unsigned short* __restrict__ WOb,
                const float* __restrict__ resid,
                const float* __restrict__ gamma,
                const float* __restrict__ beta,
                float* __restrict__ out) {
  extern __shared__ char smem[];
  unsigned short* sA = (unsigned short*)smem;  // 64 KB @ LDS offset 0
  float* sP = (float*)(smem + 65536);          // 128 KB

  const int t = threadIdx.x;
  const int wave = t >> 5, lane = t & 31;
  const int ln = lane & 15, h = lane >> 4;
  const size_t row0 = (size_t)blockIdx.x * 32;

  {  // Phase 0: O strip -> LDS via async DMA
    const char* src = (const char*)(Ob + row0 * 1024);
#pragma unroll
    for (int i = 0; i < 16; ++i) {
      unsigned off = (unsigned)(t + 256 * i) * 16u;
      async_ld_lds_b128(off, src + off);
    }
    wait_asynccnt0();
  }
  __syncthreads();

  // Phase 1: GEMM; wave owns cols [wave*128, +128)
  v8f acc[2][8];
#pragma unroll
  for (int i = 0; i < 2; ++i)
#pragma unroll
    for (int j = 0; j < 8; ++j) acc[i][j] = zero8();

#pragma clang loop unroll(disable)
  for (int kf = 0; kf < 32; ++kf) {
    opt_fence();
    FragU a[2];
#pragma unroll
    for (int mf = 0; mf < 2; ++mf) {
      const unsigned short* p = sA + (mf * 16 + ln) * 1024 + kf * 32;
      a[mf].q[0] = *(const uint4*)(p + h * 8);
      a[mf].q[1] = *(const uint4*)(p + 16 + h * 8);
    }
#pragma unroll
    for (int nf = 0; nf < 8; ++nf) {
      FragU bb;
      int d = wave * 128 + nf * 16 + ln;
      const unsigned short* p = WOb + (size_t)d * 1024 + kf * 32;
      bb.q[0] = *(const uint4*)(p + h * 8);
      bb.q[1] = *(const uint4*)(p + 16 + h * 8);
#pragma unroll
      for (int mf = 0; mf < 2; ++mf) acc[mf][nf] = wmma_bf16(a[mf], bb, acc[mf][nf]);
    }
  }
#pragma unroll
  for (int mf = 0; mf < 2; ++mf)
#pragma unroll
    for (int nf = 0; nf < 8; ++nf) {
      int col = wave * 128 + nf * 16 + ln;
#pragma unroll
      for (int v = 0; v < 8; ++v)
        sP[(mf * 16 + v + 8 * h) * 1024 + col] = acc[mf][nf][v];
    }
  __syncthreads();

  // Phase 2: residual + LayerNorm; wave handles rows [wave*4, +4)
  for (int rr = 0; rr < 4; ++rr) {
    int r = wave * 4 + rr;
    size_t g = row0 + r;
    float sum = 0.0f, sq = 0.0f;
    for (int i = 0; i < 32; ++i) {
      int c = lane + 32 * i;
      float x = sP[r * 1024 + c] + resid[g * 1024 + c];
      sum += x;
      sq += x * x;
    }
#pragma unroll
    for (int off = 16; off > 0; off >>= 1) {
      sum += __shfl_xor(sum, off, 32);
      sq += __shfl_xor(sq, off, 32);
    }
    float mu = sum * (1.0f / 1024.0f);
    float var = sq * (1.0f / 1024.0f) - mu * mu;
    float rstd = rsqrtf(var + 1e-5f);
    for (int i = 0; i < 32; ++i) {
      int c = lane + 32 * i;
      float x = sP[r * 1024 + c] + resid[g * 1024 + c];
      out[g * 1024 + c] = (x - mu) * rstd * gamma[c] + beta[c];
    }
  }
}

// ---------------------------------------------------------------------------
extern "C" void kernel_launch(void* const* d_in, const int* in_sizes, int n_in,
                              void* d_out, int out_size, void* d_ws,
                              size_t ws_size, hipStream_t stream) {
  (void)in_sizes; (void)n_in; (void)out_size; (void)ws_size;
  const float* inputQ = (const float*)d_in[0];
  const float* inputK = (const float*)d_in[1];
  const float* inputV = (const float*)d_in[2];
  const int* mask     = (const int*)d_in[3];
  const float* WQ     = (const float*)d_in[4];
  const float* WK     = (const float*)d_in[5];
  const float* WV     = (const float*)d_in[6];
  const float* WO     = (const float*)d_in[7];
  const float* gamma  = (const float*)d_in[8];
  const float* beta   = (const float*)d_in[9];
  float* out = (float*)d_out;

  // workspace layout (bf16 = u16)
  unsigned short* Qb  = (unsigned short*)d_ws;
  unsigned short* Kb  = Qb + (size_t)Mrows * 1024;
  unsigned short* Vt  = Kb + (size_t)Mrows * 1024;  // [b][col][s]
  unsigned short* Ob  = Vt + (size_t)Mrows * 1024;
  unsigned short* WOb = Ob + (size_t)Mrows * 1024;

  // allow >64KB dynamic LDS
  hipFuncSetAttribute((const void*)attn_kernel,
                      hipFuncAttributeMaxDynamicSharedMemorySize, 327680);
  hipFuncSetAttribute((const void*)outproj_ln,
                      hipFuncAttributeMaxDynamicSharedMemorySize, 196608);

  dim3 gA(Dm / 128, Mrows / 128);  // (8, 64)
  gemm_xw_bf16<false><<<gA, 256, 0, stream>>>(inputQ, WQ, Qb);
  gemm_xw_bf16<false><<<gA, 256, 0, stream>>>(inputK, WK, Kb);
  gemm_xw_bf16<true><<<gA, 256, 0, stream>>>(inputV, WV, Vt);
  cvt_f32_bf16<<<(Dm * Dm + 255) / 256, 256, 0, stream>>>(WO, WOb, Dm * Dm);
  attn_kernel<<<dim3(Sq / 32, NB), 256, 327680, stream>>>(Qb, Kb, Vt, mask, Ob);
  outproj_ln<<<dim3(Mrows / 32), 256, 196608, stream>>>(Ob, WOb, inputQ, gamma,
                                                        beta, out);
}
